// DecoderGRU_77275051589670
// MI455X (gfx1250) — matrix operand order repaired and checked by reading, not compiled, not run on hardware
//
#include <hip/hip_runtime.h>
#include <hip/hip_bf16.h>

// ---------------------------------------------------------------------------
// Persistent 2-layer GRU decoder for MI455X (gfx1250).
//  * 64 workgroups, one per 16 hidden units; each stages its 3 weight slices
//    (48 rows x 1024, bf16) into LDS once (288 KB of the 320 KB WGP LDS) and
//    keeps them there for all 1024 timesteps.
//  * All GEMMs use v_wmma_f32_16x16x32_bf16 (f32 accumulate); master hidden
//    state is kept in f32, bf16 copies feed the WMMA A-operands.
//  * Grid-wide generation barrier (device-scope atomics + s_sleep spin),
//    2 barriers per timestep; output projection fused as phase C (WGs 0..7),
//    so no (B,T,H) intermediate is ever written.
// ---------------------------------------------------------------------------

typedef __bf16 bf16_t;
typedef __attribute__((ext_vector_type(16))) __bf16        v16bf;
typedef __attribute__((ext_vector_type(8)))  float         v8f;
typedef __attribute__((ext_vector_type(4)))  unsigned int  u32x4;

namespace {
constexpr int    kB       = 64;
constexpr int    kT       = 1024;
constexpr int    kH       = 1024;
constexpr int    kO       = 128;
constexpr int    kNWG     = 64;     // one WG per 16 hidden units
constexpr int    kThreads = 128;    // 4 waves (one per 16-batch m-tile)
constexpr int    kRows    = 48;     // 3 gates x 16 hidden units
constexpr int    kRStride = 1032;   // 1024 + 8 bf16 pad -> rows hit distinct LDS bank groups
constexpr int    kMatElems = kRows * kRStride;                 // 49,536 bf16
constexpr size_t kLdsBytes = (size_t)3 * kMatElems * 2;        // 297,216 B
}

union Frag {
  u32x4 q[2];
  v16bf v;
};

__device__ __forceinline__ v8f wmma_bf16(v16bf a, v16bf b, v8f c) {
  // D = A(16x32 bf16) x B(32x16 bf16) + C(16x16 f32)
  return __builtin_amdgcn_wmma_f32_16x16x32_bf16(
      /*neg_a=*/false, a, /*neg_b=*/false, b,
      /*c_mod=*/(short)0, c, /*reuse_a=*/false, /*reuse_b=*/false);
}

__device__ __forceinline__ float sigm(float x) { return 1.0f / (1.0f + __expf(-x)); }
__device__ __forceinline__ float tanh_s(float x) {
  float e = __expf(-2.0f * fabsf(x));
  float r = (1.0f - e) / (1.0f + e);
  return copysignf(r, x);
}

__device__ __forceinline__ void grid_barrier(unsigned* bar) {
  __threadfence();
  __syncthreads();
  if (threadIdx.x == 0) {
    unsigned g = __hip_atomic_load(&bar[1], __ATOMIC_RELAXED, __HIP_MEMORY_SCOPE_AGENT);
    unsigned a = __hip_atomic_fetch_add(&bar[0], 1u, __ATOMIC_ACQ_REL, __HIP_MEMORY_SCOPE_AGENT);
    if (a == (unsigned)(kNWG - 1)) {
      __hip_atomic_store(&bar[0], 0u, __ATOMIC_RELAXED, __HIP_MEMORY_SCOPE_AGENT);
      __hip_atomic_fetch_add(&bar[1], 1u, __ATOMIC_RELEASE, __HIP_MEMORY_SCOPE_AGENT);
    } else {
      while (__hip_atomic_load(&bar[1], __ATOMIC_ACQUIRE, __HIP_MEMORY_SCOPE_AGENT) == g) {
        __builtin_amdgcn_s_sleep(1);
      }
    }
  }
  __syncthreads();
  __threadfence();
}

// ------------------------------ prep kernels -------------------------------

__global__ void cvt_bf16_kernel(const float* __restrict__ s, bf16_t* __restrict__ d, int n) {
  int i = blockIdx.x * blockDim.x + threadIdx.x;
  int stride = gridDim.x * blockDim.x;
  for (; i < n; i += stride) d[i] = (bf16_t)s[i];
}

__global__ void init_state_kernel(const float* __restrict__ z,
                                  float* __restrict__ h0f, float* __restrict__ h1f,
                                  bf16_t* __restrict__ h0b, bf16_t* __restrict__ h1b,
                                  unsigned* bar) {
  int i = blockIdx.x * blockDim.x + threadIdx.x;
  if (i == 0) { bar[0] = 0u; bar[1] = 0u; }
  if (i < kB * kH) {
    float v = z[i];
    h0f[i] = v; h1f[i] = v;
    h0b[i] = (bf16_t)v; h1b[i] = (bf16_t)v;
  }
}

// ---------------------------- persistent kernel ----------------------------

__global__ __launch_bounds__(kThreads, 1)
void gru_decoder_persistent(
    const float* __restrict__ b_ih0, const float* __restrict__ b_hh0,
    const float* __restrict__ b_ih1, const float* __restrict__ b_hh1,
    const float* __restrict__ b_outp,
    const bf16_t* __restrict__ wb_hh0, const bf16_t* __restrict__ wb_ih1,
    const bf16_t* __restrict__ wb_hh1, const bf16_t* __restrict__ wb_out,
    float* __restrict__ h0f, float* __restrict__ h1f,
    bf16_t* __restrict__ h0b, bf16_t* __restrict__ h1b,
    unsigned* bar, float* __restrict__ outp)
{
  extern __shared__ char smem_raw[];
  bf16_t* lds = (bf16_t*)smem_raw;

  const int tid   = threadIdx.x;
  const int wave  = tid >> 5;          // m-tile (16 batch rows each)
  const int lane  = tid & 31;
  const int lidx  = lane & 15;
  const int half  = lane >> 4;
  const int wg    = blockIdx.x;
  const int jbase = wg * 16;           // this WG's hidden-unit slice
  const int j     = jbase + lidx;
  const int arow  = wave * 16 + lidx;  // A-fragment source row (batch)

  // ---- stage weight slices into LDS: [0]=W_hh0, [1]=W_ih1, [2]=W_hh1 ----
  {
    const bf16_t* srcs[3] = { wb_hh0, wb_ih1, wb_hh1 };
    #pragma unroll
    for (int m = 0; m < 3; ++m) {
      const bf16_t* src = srcs[m];
      for (int lr = 0; lr < kRows; ++lr) {
        const int grow = (lr >> 4) * kH + jbase + (lr & 15);  // gate-major rows
        const u32x4* s = (const u32x4*)(src + (size_t)grow * kH);
        u32x4* d = (u32x4*)(lds + m * kMatElems + lr * kRStride);
        d[tid] = s[tid];  // 128 threads x 16B = one 1024-elem row
      }
    }
  }
  __syncthreads();

  // ---- per-lane gate biases for the fixed hidden column j ----
  const float bi0r = b_ih0[j], bi0z = b_ih0[kH + j], bi0n = b_ih0[2 * kH + j];
  const float bh0r = b_hh0[j], bh0z = b_hh0[kH + j], bh0n = b_hh0[2 * kH + j];
  const float bi1r = b_ih1[j], bi1z = b_ih1[kH + j], bi1n = b_ih1[2 * kH + j];
  const float bh1r = b_hh1[j], bh1z = b_hh1[kH + j], bh1n = b_hh1[2 * kH + j];
  const float bo   = (wg < 8) ? b_outp[wg * 16 + lidx] : 0.0f;

  const v8f vzero = {0.f, 0.f, 0.f, 0.f, 0.f, 0.f, 0.f, 0.f};

  for (int t = 0; t < kT; ++t) {
    // ================= phase A: layer 0 (input gates = b_ih_0, x==0) =====
    {
      v8f aR = vzero, aZ = vzero, aN = vzero;
      const char* aB = (const char*)(h0b + (size_t)arow * kH);
      #pragma unroll 4
      for (int ks = 0; ks < 32; ++ks) {
        Frag A;  // lanes 0-15: K {0..15}; lanes 16-31: K {16..31}
        A.q[0] = *(const u32x4*)(aB + (size_t)(ks * 32 + half * 16) * 2);
        A.q[1] = *(const u32x4*)(aB + (size_t)(ks * 32 + half * 16 + 8) * 2);
        const bf16_t* l0 = lds + lidx * kRStride + ks * 32 + half * 16;
        Frag Br, Bz, Bn;
        Br.q[0] = *(const u32x4*)(l0);
        Br.q[1] = *(const u32x4*)(l0 + 8);
        Bz.q[0] = *(const u32x4*)(l0 + 16 * kRStride);
        Bz.q[1] = *(const u32x4*)(l0 + 16 * kRStride + 8);
        Bn.q[0] = *(const u32x4*)(l0 + 32 * kRStride);
        Bn.q[1] = *(const u32x4*)(l0 + 32 * kRStride + 8);
        aR = wmma_bf16(A.v, Br.v, aR);
        aZ = wmma_bf16(A.v, Bz.v, aZ);
        aN = wmma_bf16(A.v, Bn.v, aN);
      }
      #pragma unroll
      for (int v = 0; v < 8; ++v) {
        const int b   = wave * 16 + v + 8 * half;  // C/D: M = v + 8*(lane>=16)
        const int idx = b * kH + j;
        const float hr = aR[v] + bh0r;
        const float hz = aZ[v] + bh0z;
        const float hn = aN[v] + bh0n;
        const float r  = sigm(bi0r + hr);
        const float zg = sigm(bi0z + hz);
        const float nn = tanh_s(bi0n + r * hn);
        const float hnew = (1.0f - zg) * nn + zg * h0f[idx];
        h0f[idx] = hnew;
        h0b[idx] = (bf16_t)hnew;
      }
    }
    grid_barrier(bar);

    // ================= phase B: layer 1 (x = y0_t = h0 just updated) =====
    {
      v8f xR = vzero, xZ = vzero, xN = vzero;
      v8f hR = vzero, hZ = vzero, hN = vzero;
      const char* aXB = (const char*)(h0b + (size_t)arow * kH);
      const char* aHB = (const char*)(h1b + (size_t)arow * kH);
      #pragma unroll 2
      for (int ks = 0; ks < 32; ++ks) {
        Frag Ax, Ah;
        Ax.q[0] = *(const u32x4*)(aXB + (size_t)(ks * 32 + half * 16) * 2);
        Ax.q[1] = *(const u32x4*)(aXB + (size_t)(ks * 32 + half * 16 + 8) * 2);
        Ah.q[0] = *(const u32x4*)(aHB + (size_t)(ks * 32 + half * 16) * 2);
        Ah.q[1] = *(const u32x4*)(aHB + (size_t)(ks * 32 + half * 16 + 8) * 2);
        const bf16_t* lx = lds + kMatElems     + lidx * kRStride + ks * 32 + half * 16;
        const bf16_t* lh = lds + 2 * kMatElems + lidx * kRStride + ks * 32 + half * 16;
        Frag Bx0, Bx1, Bx2, Bh0, Bh1, Bh2;
        Bx0.q[0] = *(const u32x4*)(lx);                     Bx0.q[1] = *(const u32x4*)(lx + 8);
        Bx1.q[0] = *(const u32x4*)(lx + 16 * kRStride);     Bx1.q[1] = *(const u32x4*)(lx + 16 * kRStride + 8);
        Bx2.q[0] = *(const u32x4*)(lx + 32 * kRStride);     Bx2.q[1] = *(const u32x4*)(lx + 32 * kRStride + 8);
        Bh0.q[0] = *(const u32x4*)(lh);                     Bh0.q[1] = *(const u32x4*)(lh + 8);
        Bh1.q[0] = *(const u32x4*)(lh + 16 * kRStride);     Bh1.q[1] = *(const u32x4*)(lh + 16 * kRStride + 8);
        Bh2.q[0] = *(const u32x4*)(lh + 32 * kRStride);     Bh2.q[1] = *(const u32x4*)(lh + 32 * kRStride + 8);
        xR = wmma_bf16(Ax.v, Bx0.v, xR);
        xZ = wmma_bf16(Ax.v, Bx1.v, xZ);
        xN = wmma_bf16(Ax.v, Bx2.v, xN);
        hR = wmma_bf16(Ah.v, Bh0.v, hR);
        hZ = wmma_bf16(Ah.v, Bh1.v, hZ);
        hN = wmma_bf16(Ah.v, Bh2.v, hN);
      }
      #pragma unroll
      for (int v = 0; v < 8; ++v) {
        const int b   = wave * 16 + v + 8 * half;
        const int idx = b * kH + j;
        const float xr = xR[v] + bi1r, xz = xZ[v] + bi1z, xn = xN[v] + bi1n;
        const float hr = hR[v] + bh1r, hz = hZ[v] + bh1z, hn = hN[v] + bh1n;
        const float r  = sigm(xr + hr);
        const float zg = sigm(xz + hz);
        const float nn = tanh_s(xn + r * hn);
        const float hnew = (1.0f - zg) * nn + zg * h1f[idx];
        h1f[idx] = hnew;
        h1b[idx] = (bf16_t)hnew;
      }
    }
    grid_barrier(bar);

    // ================= phase C: fused output projection for step t =======
    // out(64,128) = h1_t(64,1024) x W_out^T ; WGs 0..7 own n-tiles, waves m-tiles.
    if (wg < 8) {
      v8f acc = vzero;
      const char* aB = (const char*)(h1b + (size_t)arow * kH);
      const bf16_t* wrow = wb_out + (size_t)(wg * 16 + lidx) * kH;
      #pragma unroll 4
      for (int ks = 0; ks < 32; ++ks) {
        Frag A, Bw;
        A.q[0]  = *(const u32x4*)(aB + (size_t)(ks * 32 + half * 16) * 2);
        A.q[1]  = *(const u32x4*)(aB + (size_t)(ks * 32 + half * 16 + 8) * 2);
        Bw.q[0] = *(const u32x4*)(wrow + ks * 32 + half * 16);
        Bw.q[1] = *(const u32x4*)(wrow + ks * 32 + half * 16 + 8);
        acc = wmma_bf16(A.v, Bw.v, acc);
      }
      #pragma unroll
      for (int v = 0; v < 8; ++v) {
        const int b = wave * 16 + v + 8 * half;
        outp[(size_t)b * (kT * kO) + (size_t)t * kO + wg * 16 + lidx] = acc[v] + bo;
      }
    }
    // Next-step barrier (after phase A) separates C_t reads of h1b from
    // B_{t+1} writes; A_{t+1} touches only h0 buffers, which C never reads.
  }
}

// -------------------------------- launcher ---------------------------------

extern "C" void kernel_launch(void* const* d_in, const int* in_sizes, int n_in,
                              void* d_out, int out_size, void* d_ws, size_t ws_size,
                              hipStream_t stream) {
  (void)in_sizes; (void)n_in; (void)out_size; (void)ws_size;

  const float* z      = (const float*)d_in[0];
  // d_in[1] = seq_len (int scalar) -- compile-time constant kT = 1024
  // d_in[2] = W_ih_0 (unused: layer-0 input is all-zero)
  const float* W_hh_0 = (const float*)d_in[3];
  const float* b_ih_0 = (const float*)d_in[4];
  const float* b_hh_0 = (const float*)d_in[5];
  const float* W_ih_1 = (const float*)d_in[6];
  const float* W_hh_1 = (const float*)d_in[7];
  const float* b_ih_1 = (const float*)d_in[8];
  const float* b_hh_1 = (const float*)d_in[9];
  const float* W_out  = (const float*)d_in[10];
  const float* b_out  = (const float*)d_in[11];
  float* out = (float*)d_out;

  // workspace layout (256B-aligned slabs, ~20 MB total)
  char* ws = (char*)d_ws;
  size_t off = 0;
  auto alloc = [&](size_t bytes) {
    size_t r = off;
    off += (bytes + 255) & ~(size_t)255;
    return r;
  };
  unsigned* bar   = (unsigned*)(ws + alloc(256));
  float*    h0f   = (float*)   (ws + alloc((size_t)kB * kH * 4));
  float*    h1f   = (float*)   (ws + alloc((size_t)kB * kH * 4));
  bf16_t*   h0b   = (bf16_t*)  (ws + alloc((size_t)kB * kH * 2));
  bf16_t*   h1b   = (bf16_t*)  (ws + alloc((size_t)kB * kH * 2));
  bf16_t*   wbhh0 = (bf16_t*)  (ws + alloc((size_t)3 * kH * kH * 2));
  bf16_t*   wbih1 = (bf16_t*)  (ws + alloc((size_t)3 * kH * kH * 2));
  bf16_t*   wbhh1 = (bf16_t*)  (ws + alloc((size_t)3 * kH * kH * 2));
  bf16_t*   wbout = (bf16_t*)  (ws + alloc((size_t)kO * kH * 2));

  // prep: state init (also resets barrier words) + weight bf16 conversion
  init_state_kernel<<<(kB * kH + 255) / 256, 256, 0, stream>>>(z, h0f, h1f, h0b, h1b, bar);
  cvt_bf16_kernel<<<1024, 256, 0, stream>>>(W_hh_0, wbhh0, 3 * kH * kH);
  cvt_bf16_kernel<<<1024, 256, 0, stream>>>(W_ih_1, wbih1, 3 * kH * kH);
  cvt_bf16_kernel<<<1024, 256, 0, stream>>>(W_hh_1, wbhh1, 3 * kH * kH);
  cvt_bf16_kernel<<<128, 256, 0, stream>>>(W_out, wbout, kO * kH);

  // allow >64KB dynamic LDS (no-op if unnecessary on this stack)
  (void)hipFuncSetAttribute(reinterpret_cast<const void*>(gru_decoder_persistent),
                            hipFuncAttributeMaxDynamicSharedMemorySize, (int)kLdsBytes);

  gru_decoder_persistent<<<kNWG, kThreads, kLdsBytes, stream>>>(
      b_ih_0, b_hh_0, b_ih_1, b_hh_1, b_out,
      wbhh0, wbih1, wbhh1, wbout,
      h0f, h1f, h0b, h1b, bar, out);
}
